// HEATNet2_23957327577906
// MI455X (gfx1250) — matrix operand order repaired and checked by reading, not compiled, not run on hardware
//
#include <hip/hip_runtime.h>
#include <hip/hip_bf16.h>
#include <math.h>

#define NNODES 50000
#define NEDGES 500000
#define HID    128
#define HEADS  8
#define TTYPES 2
#define LDSPITCH 136   // 128 + 8 bf16 pad to stagger LDS banks

#if defined(__has_builtin)
# if __has_builtin(__builtin_amdgcn_global_load_async_to_lds_b128)
#  define HAVE_ASYNC_LDS 1
# endif
#endif

typedef __attribute__((ext_vector_type(16))) __bf16 v16bf;
typedef __attribute__((ext_vector_type(8)))  float  v8f;
typedef int v4i __attribute__((vector_size(16)));            // matches builtin param
typedef __attribute__((address_space(1))) v4i* gv4i_p;       // global 4xi32 vector ptr
typedef __attribute__((address_space(3))) v4i* lv4i_p;       // LDS    4xi32 vector ptr

union Frag16 {
  v16bf v;
  unsigned short u[16];
  uint4 q2[2];
};

static __device__ __forceinline__ unsigned short f2bf(float f) {
  unsigned u = __float_as_uint(f);
  u += 0x7FFFu + ((u >> 16) & 1u);            // round-to-nearest-even
  return (unsigned short)(u >> 16);
}

// monotonic float<->uint encoding so atomicMax(u32) orders signed floats
static __device__ __forceinline__ unsigned encf(float f) {
  unsigned u = __float_as_uint(f);
  return (u & 0x80000000u) ? ~u : (u | 0x80000000u);
}
static __device__ __forceinline__ float decf(unsigned u) {
  unsigned v = (u & 0x80000000u) ? (u & 0x7FFFFFFFu) : ~u;
  return __uint_as_float(v);
}

// ---------------------------------------------------------------------------
// one-shot f32 -> bf16 conversion (weights only; ~0.6 MB total)
// ---------------------------------------------------------------------------
__global__ void cvt_bf16(const float* __restrict__ src,
                         unsigned short* __restrict__ dst, long n) {
  long i = (long)blockIdx.x * blockDim.x + threadIdx.x;
  if (i < n) dst[i] = f2bf(src[i]);
}

// ---------------------------------------------------------------------------
// Y[M,128] = (xscale * X[M,128]) @ W[128,128]^T + b ; optional sigmoid-skip mix
// ABF16: A read directly as bf16 (no conversion VALU in the hot loop).
// W pre-converted to bf16; staged once per block into LDS via the gfx1250
// async-to-LDS path (fallback: plain copy); every wave computes one 16-row
// strip across all 8 column tiles with v_wmma_f32_16x16x32_bf16.
// Optionally writes a bf16 shadow copy of the output (Ybf) so downstream
// GEMMs can take the ABF16 path.
// ---------------------------------------------------------------------------
template <bool ABF16>
__global__ __launch_bounds__(256)
void wmma_gemm(const float* __restrict__ Xf,
               const unsigned short* __restrict__ Xbf,
               const unsigned short* __restrict__ Wbf,
               const float* __restrict__ bias,
               float* __restrict__ Y,
               unsigned short* __restrict__ Ybf,
               const float* __restrict__ skipFeat,
               const float* __restrict__ skipParam,
               float xscale, int M)
{
  __shared__ unsigned short lw[128 * LDSPITCH];

  // stage the 32KB bf16 weight matrix: 2048 x b128
#if HAVE_ASYNC_LDS
  {
    for (int i = threadIdx.x; i < 2048; i += 256) {
      const int row = i >> 4;        // 16 x b128 per 128-elem row
      const int c16 = i & 15;
      __builtin_amdgcn_global_load_async_to_lds_b128(
          (gv4i_p)(v4i*)((const uint4*)Wbf + i),
          (lv4i_p)(v4i*)((uint4*)(lw + row * LDSPITCH) + c16),
          0, 0);
    }
# if defined(__has_builtin) && __has_builtin(__builtin_amdgcn_s_wait_asynccnt)
    __builtin_amdgcn_s_wait_asynccnt(0);
# else
    asm volatile("s_wait_asynccnt 0x0" ::: "memory");
# endif
  }
#else
  {
    const uint4* src = (const uint4*)Wbf;
    for (int i = threadIdx.x; i < 2048; i += 256) {
      const int row = i >> 4;
      const int c16 = i & 15;
      *((uint4*)(lw + row * LDSPITCH) + c16) = src[i];
    }
  }
#endif
  __syncthreads();

  const int lane  = threadIdx.x & 31;
  const int wid   = threadIdx.x >> 5;
  const int strip = blockIdx.x * 8 + wid;
  const int half  = lane >> 4;      // 0: lanes 0-15, 1: lanes 16-31
  const int l16   = lane & 15;

  if (strip * 16 >= M) return;      // wave-uniform; EXEC stays all-ones

  v8f acc[8] = {};

  const float* Xr = ABF16 ? nullptr : (Xf + (size_t)(strip * 16 + l16) * HID);
  const unsigned short* Xrb = ABF16 ? (Xbf + (size_t)(strip * 16 + l16) * HID) : nullptr;

  for (int k = 0; k < HID; k += 32) {
    Frag16 A;
    const int ka = k + half * 8;
    if constexpr (ABF16) {
      A.q2[0] = *(const uint4*)(Xrb + ka);
      A.q2[1] = *(const uint4*)(Xrb + ka + 16);
    } else {
#pragma unroll
      for (int i = 0; i < 8; ++i) A.u[i]     = f2bf(Xr[ka + i] * xscale);
#pragma unroll
      for (int i = 0; i < 8; ++i) A.u[8 + i] = f2bf(Xr[ka + 16 + i] * xscale);
    }

#pragma unroll
    for (int nt = 0; nt < 8; ++nt) {
      Frag16 B;
      const unsigned short* wr = lw + (nt * 16 + l16) * LDSPITCH + k + half * 8;
      B.q2[0] = *((const uint4*)wr);
      B.q2[1] = *((const uint4*)wr + 2);
      acc[nt] = __builtin_amdgcn_wmma_f32_16x16x32_bf16(
          false, A.v, false, B.v, (short)0, acc[nt], false, false);
    }
  }

  float alpha = 1.0f, beta = 0.0f;
  if (skipFeat) {
    float sp = *skipParam;
    alpha = 1.0f / (1.0f + __expf(-sp));
    beta  = 1.0f - alpha;
  }

#pragma unroll
  for (int nt = 0; nt < 8; ++nt) {
    const int col = nt * 16 + l16;
    const float bb = bias[col];
#pragma unroll
    for (int r = 0; r < 8; ++r) {
      const int orow = strip * 16 + half * 8 + r;
      float val = acc[nt][r] + bb;
      if (skipFeat) val = val * alpha + skipFeat[(size_t)orow * HID + col] * beta;
      Y[(size_t)orow * HID + col] = val;
      if (Ybf) Ybf[(size_t)orow * HID + col] = f2bf(val);
    }
  }
}

// ---------------------------------------------------------------------------
// utility fills
// ---------------------------------------------------------------------------
__global__ void fill_f32(float* p, float v, long n) {
  long i = (long)blockIdx.x * blockDim.x + threadIdx.x;
  if (i < n) p[i] = v;
}
__global__ void fill_u32(unsigned* p, unsigned v, long n) {
  long i = (long)blockIdx.x * blockDim.x + threadIdx.x;
  if (i < n) p[i] = v;
}

// ---------------------------------------------------------------------------
// Pass 1: SDDMM  score[e,h] = (q[dst,h,:] . k[src,h,:]) * (sim*eW+eb) / sqrt(DK)
//         + atomic segmented max into smax[dst,h] (encoded)
// ---------------------------------------------------------------------------
__global__ __launch_bounds__(256)
void sddmm_kernel(const float* __restrict__ q, const float* __restrict__ k,
                  const int* __restrict__ src, const int* __restrict__ dst,
                  const float* __restrict__ sim,
                  const float* __restrict__ eW, const float* __restrict__ eb,
                  float* __restrict__ score, unsigned* __restrict__ smax, int E)
{
  long t = (long)blockIdx.x * blockDim.x + threadIdx.x;
  if (t >= (long)E * HEADS) return;
  const int e = (int)(t >> 3);
  const int h = (int)(t & 7);
  const int s = src[e], d = dst[e];
  const float4* qp = (const float4*)(q + (size_t)d * HID + h * 16);
  const float4* kp = (const float4*)(k + (size_t)s * HID + h * 16);
  float acc = 0.0f;
#pragma unroll
  for (int i = 0; i < 4; ++i) {
    float4 a = qp[i], b = kp[i];
    acc += a.x * b.x + a.y * b.y + a.z * b.z + a.w * b.w;
  }
  const float ea = sim[e] * eW[0] + eb[0];
  const float sc = acc * ea * 0.25f;          // 1/sqrt(16)
  score[(size_t)e * HEADS + h] = sc;
  atomicMax(&smax[(size_t)d * HEADS + h], encf(sc));
}

// ---------------------------------------------------------------------------
// Pass 2: s = exp(score - max); score <- s ; atomic ssum[dst,h] += s
// ---------------------------------------------------------------------------
__global__ __launch_bounds__(256)
void softexp_kernel(float* __restrict__ score, const int* __restrict__ dst,
                    const unsigned* __restrict__ smax, float* __restrict__ ssum, int E)
{
  long t = (long)blockIdx.x * blockDim.x + threadIdx.x;
  if (t >= (long)E * HEADS) return;
  const int e = (int)(t >> 3);
  const int h = (int)(t & 7);
  const int d = dst[e];
  const float m = decf(smax[(size_t)d * HEADS + h]);
  const float s = __expf(score[(size_t)e * HEADS + h] - m);
  score[(size_t)e * HEADS + h] = s;
  atomicAdd(&ssum[(size_t)d * HEADS + h], s);
}

// ---------------------------------------------------------------------------
// Pass 3: agg[dst, c] += v[src, c] * (s / ssum[dst, head(c)])
// ---------------------------------------------------------------------------
__global__ __launch_bounds__(256)
void scatter_kernel(const float* __restrict__ score, const float* __restrict__ ssum,
                    const float* __restrict__ v,
                    const int* __restrict__ src, const int* __restrict__ dst,
                    float* __restrict__ agg, int E)
{
  long t = (long)blockIdx.x * blockDim.x + threadIdx.x;
  if (t >= (long)E * HID) return;
  const int e = (int)(t >> 7);
  const int c = (int)(t & 127);
  const int h = c >> 4;
  const int s = src[e], d = dst[e];
  const float attn = score[(size_t)e * HEADS + h] / ssum[(size_t)d * HEADS + h];
  atomicAdd(&agg[(size_t)d * HID + c], v[(size_t)s * HID + c] * attn);
}

// ---------------------------------------------------------------------------
// column sums over a 256-row slab (coalesced), atomically merged into pooled
// ---------------------------------------------------------------------------
__global__ __launch_bounds__(256)
void pool_kernel(const float* __restrict__ X, float* __restrict__ pooled, int M)
{
  const int col = threadIdx.x & 127;
  const int rp  = threadIdx.x >> 7;          // row parity within slab
  const int rbase = blockIdx.x * 256;
  float acc = 0.0f;
  for (int r = rbase + rp; r < rbase + 256 && r < M; r += 2)
    acc += X[(size_t)r * HID + col];
  __shared__ float sh[256];
  sh[threadIdx.x] = acc;
  __syncthreads();
  if (rp == 0) atomicAdd(&pooled[col], sh[col] + sh[128 + col]);
}

// ---------------------------------------------------------------------------
// hg[o] = sum_t ( (pooledSum[t]/M) . pred_W[t,o,:] + pred_b[t,o] )
// ---------------------------------------------------------------------------
__global__ void pred_kernel(const float* __restrict__ pooled,
                            const float* __restrict__ predW,
                            const float* __restrict__ predb,
                            float* __restrict__ out, int M)
{
  const int o = threadIdx.x;
  if (o >= 4) return;
  const float invM = 1.0f / (float)M;
  float acc = 0.0f;
  for (int t = 0; t < TTYPES; ++t) {
    float s = 0.0f;
    const float* wr = predW + (size_t)(t * 4 + o) * HID;
    const float* pr = pooled + (size_t)t * HID;
    for (int c = 0; c < HID; ++c) s += pr[c] * wr[c];
    acc += predb[t * 4 + o] + s * invM;
  }
  out[o] = acc;
}

// ---------------------------------------------------------------------------
// host orchestration
// ---------------------------------------------------------------------------
struct LP {
  const float *aW, *ab, *eW, *eb, *kW, *kb, *qW, *qb, *skip, *vW, *vb;
};

extern "C" void kernel_launch(void* const* d_in, const int* in_sizes, int n_in,
                              void* d_out, int out_size, void* d_ws, size_t ws_size,
                              hipStream_t stream)
{
  (void)in_sizes; (void)n_in; (void)out_size; (void)ws_size;

  const float* feat[2] = {(const float*)d_in[0], (const float*)d_in[1]};
  const float* simp[3] = {(const float*)d_in[2], (const float*)d_in[3], (const float*)d_in[4]};
  const int* srcp[3]   = {(const int*)d_in[5], (const int*)d_in[7], (const int*)d_in[9]};
  const int* dstp[3]   = {(const int*)d_in[6], (const int*)d_in[8], (const int*)d_in[10]};

  // params flattened in JAX pytree order (dict keys sorted):
  // adapt_W, adapt_b, layers[0]{aW,ab,eW,eb,kW,kb,qW,qb,skip,vW,vb}, layers[1]{...}, pred_W, pred_b
  const float* adapt_W = (const float*)d_in[11];
  const float* adapt_b = (const float*)d_in[12];
  LP lp[2];
  int base = 13;
  for (int l = 0; l < 2; ++l) {
    lp[l].aW   = (const float*)d_in[base + 0];
    lp[l].ab   = (const float*)d_in[base + 1];
    lp[l].eW   = (const float*)d_in[base + 2];
    lp[l].eb   = (const float*)d_in[base + 3];
    lp[l].kW   = (const float*)d_in[base + 4];
    lp[l].kb   = (const float*)d_in[base + 5];
    lp[l].qW   = (const float*)d_in[base + 6];
    lp[l].qb   = (const float*)d_in[base + 7];
    lp[l].skip = (const float*)d_in[base + 8];
    lp[l].vW   = (const float*)d_in[base + 9];
    lp[l].vb   = (const float*)d_in[base + 10];
    base += 11;
  }
  const float* pred_W = (const float*)d_in[base + 0];
  const float* pred_b = (const float*)d_in[base + 1];

  // workspace carve: f32 region first, then bf16 region
  float* ws = (float*)d_ws;
  size_t off = 0;
  auto carve = [&](size_t n) { float* p = ws + off; off += n; return p; };
  const size_t NH = (size_t)NNODES * HID;
  const size_t WN = (size_t)TTYPES * HID * HID;   // elements per weight tensor
  float* cur[2]  = {carve(NH), carve(NH)};
  float* nxt[2]  = {carve(NH), carve(NH)};
  float* kbuf[2] = {carve(NH), carve(NH)};
  float* qbuf[2] = {carve(NH), carve(NH)};
  float* vbuf[2] = {carve(NH), carve(NH)};
  float* agg[2]  = {carve(NH), carve(NH)};
  float* score   = carve((size_t)NEDGES * HEADS);
  unsigned* smax = (unsigned*)carve((size_t)NNODES * HEADS);
  float* ssum    = carve((size_t)NNODES * HEADS);
  float* pooled  = carve(2 * HID);
  unsigned short* curbf[2] = {(unsigned short*)carve((NH + 1) / 2),
                              (unsigned short*)carve((NH + 1) / 2)};
  unsigned short* nxtbf[2] = {(unsigned short*)carve((NH + 1) / 2),
                              (unsigned short*)carve((NH + 1) / 2)};
  unsigned short* bfbase = (unsigned short*)carve((WN * 9 + 1) / 2);
  unsigned short* adaptWbf = bfbase;
  unsigned short* kWbf[2] = {bfbase + WN * 1, bfbase + WN * 5};
  unsigned short* qWbf[2] = {bfbase + WN * 2, bfbase + WN * 6};
  unsigned short* vWbf[2] = {bfbase + WN * 3, bfbase + WN * 7};
  unsigned short* aWbf[2] = {bfbase + WN * 4, bfbase + WN * 8};

  // pre-convert all dense weights to bf16 (tiny, once per launch)
  {
    const int cvtb = (int)((WN + 255) / 256);
    cvt_bf16<<<cvtb, 256, 0, stream>>>(adapt_W, adaptWbf, (long)WN);
    for (int l = 0; l < 2; ++l) {
      cvt_bf16<<<cvtb, 256, 0, stream>>>(lp[l].kW, kWbf[l], (long)WN);
      cvt_bf16<<<cvtb, 256, 0, stream>>>(lp[l].qW, qWbf[l], (long)WN);
      cvt_bf16<<<cvtb, 256, 0, stream>>>(lp[l].vW, vWbf[l], (long)WN);
      cvt_bf16<<<cvtb, 256, 0, stream>>>(lp[l].aW, aWbf[l], (long)WN);
    }
  }

  const int gemm_blocks = (((NNODES + 15) / 16) + 7) / 8;   // 8 waves/block
  // f32-A GEMM (converts in-register, applies xscale, optional bf16 shadow out)
  auto gemm_f32 = [&](const float* X, const unsigned short* W, const float* b, float* Y,
                      unsigned short* Ybf, const float* skipF, const float* skipP, float xs) {
    wmma_gemm<false><<<gemm_blocks, 256, 0, stream>>>(X, nullptr, W, b, Y, Ybf,
                                                      skipF, skipP, xs, NNODES);
  };
  // bf16-A GEMM (direct b128 loads of A, no conversion VALU)
  auto gemm_bf = [&](const unsigned short* Xb, const unsigned short* W, const float* b,
                     float* Y) {
    wmma_gemm<true><<<gemm_blocks, 256, 0, stream>>>(nullptr, Xb, W, b, Y, nullptr,
                                                     nullptr, nullptr, 1.0f, NNODES);
  };

  const long EH  = (long)NEDGES * HEADS;
  const long EC  = (long)NEDGES * HID;
  const long NHl = (long)NH;
  const long NHd = (long)NNODES * HEADS;
  const int blkEH  = (int)((EH + 255) / 256);
  const int blkEC  = (int)((EC + 255) / 256);
  const int blkNH  = (int)((NHl + 255) / 256);
  const int blkNHd = (int)((NHd + 255) / 256);
  const unsigned ENC_NEG_INF = 0x007FFFFFu;   // encf(-inf)

  // input adaptation: f32 feats in, f32 cur + bf16 shadow out
  for (int t = 0; t < TTYPES; ++t)
    gemm_f32(feat[t], adaptWbf + (size_t)t * HID * HID, adapt_b + (size_t)t * HID,
             cur[t], curbf[t], nullptr, nullptr, 1.0f);

  const int st_of[3] = {0, 0, 1};
  const int dt_of[3] = {0, 1, 0};
  const float inv_cnt[2] = {0.5f, 1.0f};   // dst type 0 receives 2 etypes, type 1 one

  for (int l = 0; l < 2; ++l) {
    const LP& P = lp[l];
    // k/q/v once per node type from the bf16 activation shadow
    for (int t = 0; t < TTYPES; ++t) {
      gemm_bf(curbf[t], kWbf[l] + (size_t)t * HID * HID, P.kb + (size_t)t * HID, kbuf[t]);
      gemm_bf(curbf[t], qWbf[l] + (size_t)t * HID * HID, P.qb + (size_t)t * HID, qbuf[t]);
      gemm_bf(curbf[t], vWbf[l] + (size_t)t * HID * HID, P.vb + (size_t)t * HID, vbuf[t]);
    }
    for (int t = 0; t < TTYPES; ++t)
      fill_f32<<<blkNH, 256, 0, stream>>>(agg[t], 0.0f, NHl);

    for (int et = 0; et < 3; ++et) {
      const int st = st_of[et], dt = dt_of[et];
      fill_u32<<<blkNHd, 256, 0, stream>>>(smax, ENC_NEG_INF, NHd);
      fill_f32<<<blkNHd, 256, 0, stream>>>(ssum, 0.0f, NHd);
      sddmm_kernel<<<blkEH, 256, 0, stream>>>(qbuf[dt], kbuf[st], srcp[et], dstp[et],
                                              simp[et], P.eW, P.eb, score, smax, NEDGES);
      softexp_kernel<<<blkEH, 256, 0, stream>>>(score, dstp[et], smax, ssum, NEDGES);
      scatter_kernel<<<blkEC, 256, 0, stream>>>(score, ssum, vbuf[st], srcp[et], dstp[et],
                                                agg[dt], NEDGES);
    }

    // fused: nxt = sigmoid(skip)*( (agg/cnt) @ aW^T + ab ) + (1-sigmoid(skip))*cur
    for (int t = 0; t < TTYPES; ++t)
      gemm_f32(agg[t], aWbf[l] + (size_t)t * HID * HID, P.ab + (size_t)t * HID,
               nxt[t], nxtbf[t], cur[t], P.skip + t, inv_cnt[t]);

    for (int t = 0; t < TTYPES; ++t) {
      float* tf = cur[t]; cur[t] = nxt[t]; nxt[t] = tf;
      unsigned short* tb = curbf[t]; curbf[t] = nxtbf[t]; nxtbf[t] = tb;
    }
  }

  // readout
  fill_f32<<<1, 256, 0, stream>>>(pooled, 0.0f, 2 * HID);
  const int poolb = (NNODES + 255) / 256;
  pool_kernel<<<poolb, 256, 0, stream>>>(cur[0], pooled, NNODES);
  pool_kernel<<<poolb, 256, 0, stream>>>(cur[1], pooled + HID, NNODES);
  pred_kernel<<<1, 32, 0, stream>>>(pooled, pred_W, pred_b, (float*)d_out, NNODES);
}
